// KVCompressedGPT2Attention_17669495455862
// MI455X (gfx1250) — matrix operand order repaired and checked by reading, not compiled, not run on hardware
//
#include <hip/hip_runtime.h>
#include <hip/hip_bf16.h>
#include <math.h>

typedef __attribute__((ext_vector_type(16))) __bf16 bf16x16;
typedef __attribute__((ext_vector_type(8)))  __bf16 bf16x8;
typedef __attribute__((ext_vector_type(8)))  float  v8f;
typedef unsigned int u32x4 __attribute__((ext_vector_type(4)));
typedef int          i32x4 __attribute__((ext_vector_type(4)));
typedef int          i32x8 __attribute__((ext_vector_type(8)));

constexpr int NB = 2, NT = 2048, NC = 1024, NH = 16, ND = 64, NR = 32;
constexpr int MROWS = NB * NT;      // 4096
constexpr int NQKV  = 3 * NC;       // 3072

__device__ inline bf16x16 cat16(bf16x8 lo, bf16x8 hi) {
  return __builtin_shufflevector(lo, hi, 0,1,2,3,4,5,6,7,8,9,10,11,12,13,14,15);
}
__device__ inline v8f zero8() {
  v8f z;
#pragma unroll
  for (int i = 0; i < 8; i++) z[i] = 0.0f;
  return z;
}

#if __has_builtin(__builtin_amdgcn_tensor_load_to_lds)
#define USE_TDM 1
#else
#define USE_TDM 0
#endif

#if USE_TDM
// DMA a 128-row x 32-col bf16 tile (row stride = stride_elems) from global into
// LDS at byte offset lds_off, padding each 64B row with 16B (-> LDS row stride 40 elems).
// D# layout per CDNA5 ISA ch.8: group0 = {flags, lds_addr, global_addr, type=2},
// group1 = {mask/data_size/pad, tensor dims, tile dims, dim0 stride}.
__device__ __forceinline__ void tdm_tile_128x32(unsigned lds_off, const void* g,
                                                unsigned stride_elems,
                                                unsigned tdim0, unsigned tdim1) {
  unsigned long long ga = (unsigned long long)(uintptr_t)g;
  u32x4 g0 = { 1u,                                   // count=1, user descriptor
               lds_off,                              // lds_addr (bytes)
               (unsigned)ga,                         // global_addr[31:0]
               (unsigned)((ga >> 32) & 0x01FFFFFFu) | (2u << 30) };  // addr[56:32] | type=2
  i32x8 g1;
  g1[0] = (int)((1u << 16)      // data_size = 1 -> 2 bytes
              | (1u << 20)      // pad_enable
              | (3u << 22)      // pad_interval: 16 DWORDs (64B) between pads
              | (3u << 25));    // pad_amount: 4 DWORDs (16B)
  g1[1] = (int)((tdim0 & 0xFFFFu) << 16);                              // tensor_dim0 lo
  g1[2] = (int)(((tdim0 >> 16) & 0xFFFFu) | ((tdim1 & 0xFFFFu) << 16)); // dim0 hi | dim1 lo
  g1[3] = (int)(((tdim1 >> 16) & 0xFFFFu) | (32u << 16));              // dim1 hi | tile_dim0=32
  g1[4] = 128;                  // tile_dim1 = 128 rows (tile_dim2 = 0)
  g1[5] = (int)stride_elems;    // tensor_dim0_stride lo
  g1[6] = 0;                    // stride hi / dim1_stride lo (unused for 2D)
  g1[7] = 0;
  i32x4 z4 = {0, 0, 0, 0};
#if __has_include(<hip/amd_detail/amd_gfx1250_TDM.h>)
  i32x8 z8 = {0, 0, 0, 0, 0, 0, 0, 0};
  __builtin_amdgcn_tensor_load_to_lds(g0, g1, z4, z4, z8, 0);
#else
  __builtin_amdgcn_tensor_load_to_lds(g0, g1, z4, z4, 0);
#endif
}
__device__ __forceinline__ void wait_tensor0() {
#if __has_builtin(__builtin_amdgcn_s_wait_tensorcnt)
  __builtin_amdgcn_s_wait_tensorcnt(0);
#else
  asm volatile("s_wait_tensorcnt 0x0" ::: "memory");
#endif
}
#endif

// ---------------- small prep kernels ----------------

// out[e*64+d] = sum_r wc[e*32+r] * we[r*64+d]   (Weff = wc @ we, [64][64])
__global__ void weff_kernel(const float* __restrict__ wc, const float* __restrict__ we,
                            float* __restrict__ out) {
  int i = blockIdx.x * blockDim.x + threadIdx.x;
  if (i >= 64 * 64) return;
  int e = i >> 6, d = i & 63;
  float s = 0.f;
#pragma unroll
  for (int r = 0; r < NR; r++) s += wc[e * NR + r] * we[r * 64 + d];
  out[i] = s;
}

// Fold Weff into K/V column blocks of c_attn_w; emit transposed bf16 W' [3072][1024] + folded bias.
__global__ void fold_qkv_kernel(const float* __restrict__ W, const float* __restrict__ bin,
                                const float* __restrict__ weffk, const float* __restrict__ weffv,
                                __bf16* __restrict__ Wt, float* __restrict__ bout) {
  int i = blockIdx.x * blockDim.x + threadIdx.x;   // over 3072*1024
  int n = i >> 10, k = i & (NC - 1);
  float w;
  if (n < NC) {
    w = W[(size_t)k * NQKV + n];
    if (k == 0) bout[n] = bin[n];
  } else {
    const float* weff = (n < 2 * NC) ? weffk : weffv;
    int base = (n < 2 * NC) ? NC : 2 * NC;
    int d = n - base;
    int h = d >> 6, dd = d & 63;
    const float* wr = W + (size_t)k * NQKV + base + h * ND;
    float s = 0.f;
    for (int e = 0; e < ND; e++) s += wr[e] * weff[e * ND + dd];
    w = s;
    if (k == 0) {
      const float* br = bin + base + h * ND;
      float sb = 0.f;
      for (int e = 0; e < ND; e++) sb += br[e] * weff[e * ND + dd];
      bout[n] = sb;
    }
  }
  Wt[(size_t)n * NC + k] = (__bf16)w;
}

__global__ void cvt_bf16_kernel(const float* __restrict__ in, __bf16* __restrict__ out, int n) {
  int i = blockIdx.x * blockDim.x + threadIdx.x;
  if (i < n) out[i] = (__bf16)in[i];
}

// in[k][n] (rows x cols, row-major) -> out[n][k] bf16
__global__ void transpose_bf16_kernel(const float* __restrict__ in, __bf16* __restrict__ out,
                                      int rows, int cols) {
  int i = blockIdx.x * blockDim.x + threadIdx.x;
  if (i >= rows * cols) return;
  int k = i / cols, n = i % cols;
  out[(size_t)n * rows + k] = (__bf16)in[i];
}

// ---------------- WMMA GEMM: C[M,N] = A[M,K] @ Bt[N,K]^T + bias ----------------
// LDS layout (dynamic, 40KB): A tiles [2][128][40] bf16 @ 0, B tiles [2][128][40] @ 20480.
// mode 0: scatter into Q/K/V head-major [B,H,T,D] bf16;  mode 1: fp32 out + bias.
__global__ __launch_bounds__(256) void gemm_bf16(
    const __bf16* __restrict__ A, const __bf16* __restrict__ Bt,
    const float* __restrict__ bias, int M, int N, int K, int mode,
    __bf16* __restrict__ qout, __bf16* __restrict__ kout, __bf16* __restrict__ vout,
    float* __restrict__ fout) {
  extern __shared__ char smem[];
  constexpr int TILE_BYTES = 128 * 40 * 2;   // 10240
  const int tid = threadIdx.x;
  const int wid = tid >> 5;
  const int lane = tid & 31;
  const int lr = lane & 15;
  const int g8 = (lane >> 4) * 8;    // A-operand K-chunk select
  const int kb = (lane >> 4) * 16;   // unused; B loads use g8 like A
  const int bm0 = blockIdx.y * 128;
  const int bn0 = blockIdx.x * 128;
  const int wm = (wid & 3) * 32;     // 4 waves along M
  const int wn = (wid >> 2) * 64;    // 2 waves along N

  v8f acc[2][4];
#pragma unroll
  for (int mt = 0; mt < 2; mt++)
#pragma unroll
    for (int nt = 0; nt < 4; nt++) acc[mt][nt] = zero8();

  auto compute_step = [&](const __bf16* La, const __bf16* Lb) {
    bf16x16 av[2], bv[4];
#pragma unroll
    for (int mt = 0; mt < 2; mt++) {
      const __bf16* p = La + (wm + mt * 16 + lr) * 40;
      av[mt] = cat16(*(const bf16x8*)(p + g8), *(const bf16x8*)(p + 16 + g8));
    }
#pragma unroll
    for (int nt = 0; nt < 4; nt++) {
      const __bf16* p = Lb + (wn + nt * 16 + lr) * 40;
      bv[nt] = cat16(*(const bf16x8*)(p + g8), *(const bf16x8*)(p + 16 + g8));
    }
#pragma unroll
    for (int mt = 0; mt < 2; mt++)
#pragma unroll
      for (int nt = 0; nt < 4; nt++)
        acc[mt][nt] = __builtin_amdgcn_wmma_f32_16x16x32_bf16(
            false, av[mt], false, bv[nt], (short)0, acc[mt][nt], false, false);
  };

  const int nIter = K / 32;
#if USE_TDM
  // Double-buffered TDM pipeline: wave 0 drives the DMA engine; DMA of tile i+1
  // overlaps WMMA on tile i. One barrier per iteration.
  const bool issuer = (wid == 0);
  if (issuer) {
    tdm_tile_128x32(0u,              A  + (size_t)bm0 * K, (unsigned)K, (unsigned)K, (unsigned)M);
    tdm_tile_128x32(2u * TILE_BYTES, Bt + (size_t)bn0 * K, (unsigned)K, (unsigned)K, (unsigned)N);
  }
  for (int it = 0; it < nIter; ++it) {
    const int buf = it & 1;
    if (issuer) wait_tensor0();      // current buffer's 2 DMAs complete
    __syncthreads();                 // data visible; prev compute done -> other buf free
    if (issuer && it + 1 < nIter) {
      const int k0 = (it + 1) * 32;
      tdm_tile_128x32((unsigned)((buf ^ 1) * TILE_BYTES),
                      A + (size_t)bm0 * K + k0, (unsigned)K, (unsigned)K, (unsigned)M);
      tdm_tile_128x32((unsigned)(2 * TILE_BYTES + (buf ^ 1) * TILE_BYTES),
                      Bt + (size_t)bn0 * K + k0, (unsigned)K, (unsigned)K, (unsigned)N);
    }
    compute_step((const __bf16*)(smem + buf * TILE_BYTES),
                 (const __bf16*)(smem + 2 * TILE_BYTES + buf * TILE_BYTES));
  }
#else
  // Fallback: synchronous VGPR staging, single buffer.
  __bf16* sA = (__bf16*)smem;
  __bf16* sB = (__bf16*)(smem + 2 * TILE_BYTES);
  const int ldrow = tid >> 1;
  const int ldcol = (tid & 1) * 16;
  for (int it = 0; it < nIter; ++it) {
    const int k0 = it * 32;
    __syncthreads();
    {
      const __bf16* pa = A + (size_t)(bm0 + ldrow) * K + k0 + ldcol;
      *(bf16x8*)(sA + ldrow * 40 + ldcol)     = *(const bf16x8*)pa;
      *(bf16x8*)(sA + ldrow * 40 + ldcol + 8) = *(const bf16x8*)(pa + 8);
      const __bf16* pb = Bt + (size_t)(bn0 + ldrow) * K + k0 + ldcol;
      *(bf16x8*)(sB + ldrow * 40 + ldcol)     = *(const bf16x8*)pb;
      *(bf16x8*)(sB + ldrow * 40 + ldcol + 8) = *(const bf16x8*)(pb + 8);
    }
    __syncthreads();
    compute_step(sA, sB);
  }
#endif

  const int rsel = (lane >> 4) * 8;   // C-layout: lanes 16-31 hold rows g+8
#pragma unroll
  for (int mt = 0; mt < 2; mt++)
#pragma unroll
    for (int nt = 0; nt < 4; nt++)
#pragma unroll
      for (int gg = 0; gg < 8; gg++) {
        int m = bm0 + wm + mt * 16 + gg + rsel;
        int n = bn0 + wn + nt * 16 + lr;
        float val = acc[mt][nt][gg] + bias[n];
        if (mode == 0) {
          int part = n >> 10;
          int c = n & (NC - 1);
          int h = c >> 6, d = c & 63;
          int b = m >> 11, t = m & (NT - 1);
          size_t off = (((size_t)b * NH + h) * NT + t) * ND + d;
          __bf16 bvv = (__bf16)val;
          if (part == 0)      qout[off] = bvv;
          else if (part == 1) kout[off] = bvv;
          else                vout[off] = bvv;
        } else {
          fout[(size_t)m * N + n] = val;
        }
      }
}

// ---------------- Flash attention (causal), per (b,h), 128-query blocks ----------------
__global__ __launch_bounds__(256) void flash_attn(
    const __bf16* __restrict__ Q, const __bf16* __restrict__ Kx,
    const __bf16* __restrict__ Vx, __bf16* __restrict__ attn) {
  __shared__ __bf16 lK[64][72];        // [key][d]
  __shared__ __bf16 lVt[64][72];       // [d][key]
  __shared__ __bf16 lP[8][16][72];     // per-wave probs [row][key]
  const int tid = threadIdx.x, wid = tid >> 5, lane = tid & 31;
  const int lr = lane & 15;
  const int g8 = (lane >> 4) * 8;
  const int kb = (lane >> 4) * 16;
  const int rsel = (lane >> 4) * 8;
  const int bh = blockIdx.y;
  const int q0 = blockIdx.x * 128;
  const __bf16* Qb = Q + (size_t)bh * NT * ND;
  const __bf16* Kb = Kx + (size_t)bh * NT * ND;
  const __bf16* Vb = Vx + (size_t)bh * NT * ND;

  // Q rows for this wave, loaded once into A-operand registers (2 k-steps over D=64)
  bf16x16 qa[2];
  {
    int qr = q0 + wid * 16 + lr;
#pragma unroll
    for (int ks = 0; ks < 2; ks++) {
      const __bf16* p = Qb + (size_t)qr * ND + ks * 32 + g8;
      qa[ks] = cat16(*(const bf16x8*)p, *(const bf16x8*)(p + 16));
    }
  }
  float mrow[8], lrw[8];
  v8f oacc[4];
#pragma unroll
  for (int g = 0; g < 8; g++) { mrow[g] = -3.0e38f; lrw[g] = 0.f; }
#pragma unroll
  for (int nt = 0; nt < 4; nt++) oacc[nt] = zero8();

  const int kend = q0 + 128;
  for (int k0 = 0; k0 < kend; k0 += 64) {
    __syncthreads();
    {   // cooperative K / V-transposed tile staging
      int key = tid >> 2;
      int seg = (tid & 3) * 16;
      const __bf16* pk = Kb + (size_t)(k0 + key) * ND + seg;
      *(bf16x8*)&lK[key][seg]     = *(const bf16x8*)pk;
      *(bf16x8*)&lK[key][seg + 8] = *(const bf16x8*)(pk + 8);
      const __bf16* pv = Vb + (size_t)(k0 + key) * ND + seg;
      bf16x8 v0 = *(const bf16x8*)pv;
      bf16x8 v1 = *(const bf16x8*)(pv + 8);
#pragma unroll
      for (int j = 0; j < 8; j++) {
        lVt[seg + j][key]     = v0[j];
        lVt[seg + 8 + j][key] = v1[j];
      }
      if (k0 + 64 < kend) {   // warm L2/WGP$ for the next key block
        __builtin_prefetch(Kb + (size_t)(k0 + 64 + key) * ND + seg, 0, 1);
        __builtin_prefetch(Vb + (size_t)(k0 + 64 + key) * ND + seg, 0, 1);
      }
    }
    __syncthreads();

    // scores S = Q @ K^T  (16x64 per wave)
    v8f s[4];
#pragma unroll
    for (int nt = 0; nt < 4; nt++) s[nt] = zero8();
#pragma unroll
    for (int nt = 0; nt < 4; nt++)
#pragma unroll
      for (int ks = 0; ks < 2; ks++) {
        const __bf16* p = &lK[nt * 16 + lr][ks * 32 + g8];
        bf16x16 bk = cat16(*(const bf16x8*)p, *(const bf16x8*)(p + 16));
        s[nt] = __builtin_amdgcn_wmma_f32_16x16x32_bf16(
            false, qa[ks], false, bk, (short)0, s[nt], false, false);
      }

    // online softmax per row (rows g / g+8 split across lane halves)
#pragma unroll
    for (int gg = 0; gg < 8; gg++) {
      int qr = q0 + wid * 16 + gg + rsel;
      float sv[4];
      float rmax = -3.0e38f;
#pragma unroll
      for (int nt = 0; nt < 4; nt++) {
        int key = k0 + nt * 16 + lr;
        float x = s[nt][gg] * 0.125f;           // 1/sqrt(64)
        x = (key <= qr) ? x : -3.0e38f;         // causal mask
        sv[nt] = x;
        rmax = fmaxf(rmax, x);
      }
      rmax = fmaxf(rmax, __shfl_xor(rmax, 1));
      rmax = fmaxf(rmax, __shfl_xor(rmax, 2));
      rmax = fmaxf(rmax, __shfl_xor(rmax, 4));
      rmax = fmaxf(rmax, __shfl_xor(rmax, 8));
      float mnew = fmaxf(mrow[gg], rmax);
      float sf = __expf(mrow[gg] - mnew);
      mrow[gg] = mnew;
      float rsum = 0.f;
#pragma unroll
      for (int nt = 0; nt < 4; nt++) {
        float pv_ = __expf(sv[nt] - mnew);
        sv[nt] = pv_;
        rsum += pv_;
      }
      rsum += __shfl_xor(rsum, 1);
      rsum += __shfl_xor(rsum, 2);
      rsum += __shfl_xor(rsum, 4);
      rsum += __shfl_xor(rsum, 8);
      lrw[gg] = lrw[gg] * sf + rsum;
#pragma unroll
      for (int nt = 0; nt < 4; nt++) oacc[nt][gg] *= sf;
#pragma unroll
      for (int nt = 0; nt < 4; nt++)
        lP[wid][gg + rsel][nt * 16 + lr] = (__bf16)sv[nt];
    }
    __syncthreads();   // make P visible (and keep barrier count block-uniform)

    // O += P @ V   (P re-read from LDS in A-operand layout)
#pragma unroll
    for (int ks = 0; ks < 2; ks++) {
      const __bf16* pp = &lP[wid][lr][0];
      bf16x16 pa = cat16(*(const bf16x8*)(pp + ks * 32 + g8),
                         *(const bf16x8*)(pp + ks * 32 + 16 + g8));
#pragma unroll
      for (int nt = 0; nt < 4; nt++) {
        const __bf16* pv = &lVt[nt * 16 + lr][ks * 32 + g8];
        bf16x16 bv = cat16(*(const bf16x8*)pv, *(const bf16x8*)(pv + 16));
        oacc[nt] = __builtin_amdgcn_wmma_f32_16x16x32_bf16(
            false, pa, false, bv, (short)0, oacc[nt], false, false);
      }
    }
  }

  // normalize + merge-heads store: attn[b][t][h*64+d] bf16
  const int b = bh >> 4, h = bh & 15;
#pragma unroll
  for (int nt = 0; nt < 4; nt++)
#pragma unroll
    for (int gg = 0; gg < 8; gg++) {
      int t = q0 + wid * 16 + gg + rsel;
      float val = oacc[nt][gg] / lrw[gg];
      attn[((size_t)(b * NT + t)) * NC + h * ND + nt * 16 + lr] = (__bf16)val;
    }
}

// ---------------- launcher ----------------
extern "C" void kernel_launch(void* const* d_in, const int* in_sizes, int n_in,
                              void* d_out, int out_size, void* d_ws, size_t ws_size,
                              hipStream_t stream) {
  const float* hidden   = (const float*)d_in[0];
  const float* c_attn_w = (const float*)d_in[1];
  const float* c_attn_b = (const float*)d_in[2];
  const float* c_proj_w = (const float*)d_in[3];
  const float* c_proj_b = (const float*)d_in[4];
  const float* wk_c     = (const float*)d_in[5];
  const float* wk_e     = (const float*)d_in[6];
  const float* wv_c     = (const float*)d_in[7];
  const float* wv_e     = (const float*)d_in[8];
  (void)in_sizes; (void)n_in; (void)out_size; (void)ws_size;

  char* ws = (char*)d_ws;
  size_t off = 0;
  auto alloc = [&](size_t bytes) -> void* {
    void* p = ws + off;
    off += (bytes + 255) & ~(size_t)255;
    return p;
  };
  float*  weffk  = (float*)alloc(64 * 64 * 4);
  float*  weffv  = (float*)alloc(64 * 64 * 4);
  __bf16* WqkvT  = (__bf16*)alloc((size_t)NQKV * NC * 2);
  float*  bqkv   = (float*)alloc(NQKV * 4);
  __bf16* Xb     = (__bf16*)alloc((size_t)MROWS * NC * 2);
  __bf16* Qh     = (__bf16*)alloc((size_t)NB * NH * NT * ND * 2);
  __bf16* Kh     = (__bf16*)alloc((size_t)NB * NH * NT * ND * 2);
  __bf16* Vh     = (__bf16*)alloc((size_t)NB * NH * NT * ND * 2);
  __bf16* attnb  = (__bf16*)alloc((size_t)MROWS * NC * 2);
  __bf16* WprojT = (__bf16*)alloc((size_t)NC * NC * 2);

  weff_kernel<<<16, 256, 0, stream>>>(wk_c, wk_e, weffk);
  weff_kernel<<<16, 256, 0, stream>>>(wv_c, wv_e, weffv);
  fold_qkv_kernel<<<(NQKV * NC) / 256, 256, 0, stream>>>(
      c_attn_w, c_attn_b, weffk, weffv, WqkvT, bqkv);
  cvt_bf16_kernel<<<(MROWS * NC) / 256, 256, 0, stream>>>(hidden, Xb, MROWS * NC);
  transpose_bf16_kernel<<<(NC * NC) / 256, 256, 0, stream>>>(c_proj_w, WprojT, NC, NC);

  constexpr int GEMM_LDS = 4 * 128 * 40 * 2;   // 40960 bytes
  gemm_bf16<<<dim3(NQKV / 128, MROWS / 128), 256, GEMM_LDS, stream>>>(
      Xb, WqkvT, bqkv, MROWS, NQKV, NC, 0, Qh, Kh, Vh, nullptr);

  flash_attn<<<dim3(NT / 128, NB * NH), 256, 0, stream>>>(Qh, Kh, Vh, attnb);

  gemm_bf16<<<dim3(NC / 128, MROWS / 128), 256, GEMM_LDS, stream>>>(
      attnb, WprojT, c_proj_b, MROWS, NC, NC, 1, nullptr, nullptr, nullptr,
      (float*)d_out);
}
